// MultiHeadAttentionLayer_43104291783093
// MI455X (gfx1250) — matrix operand order repaired and checked by reading, not compiled, not run on hardware
//
#include <hip/hip_runtime.h>

// Problem constants (fixed by the reference)
#define N_B 4
#define L_S 2048
#define D_M 1024
#define H_N 16
#define HD  64

typedef __attribute__((ext_vector_type(16))) _Float16 v16h;
typedef __attribute__((ext_vector_type(8)))  _Float16 v8h;
typedef __attribute__((ext_vector_type(4)))  _Float16 v4h;
typedef __attribute__((ext_vector_type(8)))  float    v8f;
typedef __attribute__((ext_vector_type(4)))  float    v4f;

__device__ __forceinline__ v8f vzero8() {
  v8f v;
#pragma unroll
  for (int i = 0; i < 8; ++i) v[i] = 0.0f;
  return v;
}

__device__ __forceinline__ v8f wmma_f16(v16h a, v16h b, v8f c) {
  // (neg_a, A, neg_b, B, c_mod, C, reuse_a, reuse_b)
  return __builtin_amdgcn_wmma_f32_16x16x32_f16(false, a, false, b, (short)0, c,
                                                false, false);
}

__device__ __forceinline__ v16h cat8(v8h lo, v8h hi) {
  return __builtin_shufflevector(lo, hi, 0, 1, 2, 3, 4, 5, 6, 7, 8, 9, 10, 11,
                                 12, 13, 14, 15);
}

// A fragment: 16(M) x 32(K) f16 tile, row-major, row stride ld (halves).
// lane<16: row=lane, K chunks {0..7} and {16..23}; lane>=16: row=lane-16,
// K chunks {8..15} and {24..31}.   (ISA 7.12.2, 16-bit A 16x32)
__device__ __forceinline__ v16h load_a(const _Float16* __restrict__ base,
                                       int ld) {
  const int lane = threadIdx.x & 31;
  const _Float16* p = base + (lane & 15) * ld + ((lane >> 4) << 3);
  v8h lo = *reinterpret_cast<const v8h*>(p);
  v8h hi = *reinterpret_cast<const v8h*>(p + 16);
  return cat8(lo, hi);
}

// B fragment: 32(K) x 16(N) B operand, loaded from B^T stored row-major
// (16 rows = N, 32 cols = K), row stride ld (halves).
// lanes 0-15 hold K={0..7,16..23} of column n=lane; lanes 16-31 hold K={8..15,24..31}.
__device__ __forceinline__ v16h load_b(const _Float16* __restrict__ base,
                                       int ld) {
  const int lane = threadIdx.x & 31;
  const _Float16* p = base + (lane & 15) * ld + ((lane >> 4) << 3);
  v8h lo = *reinterpret_cast<const v8h*>(p);
  v8h hi = *reinterpret_cast<const v8h*>(p + 16);
  return cat8(lo, hi);
}

// Per-lane prefetch of a fragment's cacheline -> global_prefetch_b8
// locality 3 => nearest-scope prefetch (fills WGP cache + GL2)
__device__ __forceinline__ void pf_frag(const _Float16* base, int ld) {
  const int lane = threadIdx.x & 31;
  __builtin_prefetch((const void*)(base + (lane & 15) * ld), 0, 3);
}

__device__ __forceinline__ float half_max16(float v) {
  v = fmaxf(v, __shfl_xor(v, 1, 32));
  v = fmaxf(v, __shfl_xor(v, 2, 32));
  v = fmaxf(v, __shfl_xor(v, 4, 32));
  v = fmaxf(v, __shfl_xor(v, 8, 32));
  return v;
}

__device__ __forceinline__ float half_sum16(float v) {
  v += __shfl_xor(v, 1, 32);
  v += __shfl_xor(v, 2, 32);
  v += __shfl_xor(v, 4, 32);
  v += __shfl_xor(v, 8, 32);
  return v;
}

// ---------------------------------------------------------------- convert
__global__ void __launch_bounds__(256) cvt_f32_f16(
    const float* __restrict__ in, _Float16* __restrict__ out, int n4) {
  int i = blockIdx.x * blockDim.x + threadIdx.x;
  if (i < n4) {
    v4f v = *reinterpret_cast<const v4f*>(in + 4 * (size_t)i);
    v4h o;
#pragma unroll
    for (int k = 0; k < 4; ++k) o[k] = (_Float16)v[k];
    *reinterpret_cast<v4h*>(out + 4 * (size_t)i) = o;
  }
}

// ------------------------------------------------------ GEMM  Y = X*W^T + b
// X: 8192x1024 f16 row-major.  Wt: 1024x1024 f16 row-major (== W, acts as
// B^T of W^T).  One wave computes a 32x64 output tile.  Parity ping-pong
// software pipeline: step s+1's fragments are loaded (into the other buffer
// bank) while step s's WMMAs execute -- no rotation moves, partial waits.
// mode 0: Yh row-major f16       (Q, K projections)
// mode 1: Yh = Vt[n][h][d][t]    (V projection, per-head transposed)
// mode 2: Yf row-major f32       (final output projection)
__global__ void __launch_bounds__(128) gemm_wmma_32x64(
    const _Float16* __restrict__ X, const _Float16* __restrict__ Wt,
    const float* __restrict__ bias, _Float16* __restrict__ Yh,
    float* __restrict__ Yf, int mode) {
  const int wave = blockIdx.x * (blockDim.x >> 5) + (threadIdx.x >> 5);
  const int tiles_n = D_M / 64;  // 16
  const int row0 = (wave / tiles_n) * 32;
  const int col0 = (wave % tiles_n) * 64;
  const int lane = threadIdx.x & 31;

  const _Float16* xb0 = X + (size_t)row0 * D_M;
  const _Float16* xb1 = X + (size_t)(row0 + 16) * D_M;
  const _Float16* wb = Wt + (size_t)col0 * D_M;

  v8f acc[2][4];
#pragma unroll
  for (int i = 0; i < 2; ++i)
#pragma unroll
    for (int j = 0; j < 4; ++j) acc[i][j] = vzero8();

  v16h ab[2][2];  // [parity][row half]
  v16h bb[2][4];  // [parity][col tile]

  // prologue: fragments for k-step 0 into bank 0
  ab[0][0] = load_a(xb0, D_M);
  ab[0][1] = load_a(xb1, D_M);
#pragma unroll
  for (int j = 0; j < 4; ++j)
    bb[0][j] = load_b(wb + (size_t)(16 * j) * D_M, D_M);

  // one k-step: issue next-step loads into the opposite bank, then WMMA
  auto gstep = [&](int s, int p) {
    if (s < 31) {
      const int kn = (s + 1) << 5;
      pf_frag(xb0 + kn + 128, D_M);  // ~4 steps ahead
      pf_frag(xb1 + kn + 128, D_M);
      pf_frag(wb + kn + 128, D_M);
      pf_frag(wb + (size_t)32 * D_M + kn + 128, D_M);
      ab[p ^ 1][0] = load_a(xb0 + kn, D_M);
      ab[p ^ 1][1] = load_a(xb1 + kn, D_M);
#pragma unroll
      for (int j = 0; j < 4; ++j)
        bb[p ^ 1][j] = load_b(wb + (size_t)(16 * j) * D_M + kn, D_M);
    }
#pragma unroll
    for (int j = 0; j < 4; ++j)
      acc[0][j] = wmma_f16(ab[p][0], bb[p][j], acc[0][j]);
#pragma unroll
    for (int j = 0; j < 4; ++j)
      acc[1][j] = wmma_f16(ab[p][1], bb[p][j], acc[1][j]);
  };

  for (int s = 0; s < 32; s += 2) {  // 1024/32 = 32 k-steps
    gstep(s, 0);
    gstep(s + 1, 1);
  }

  const int half8 = (lane >> 4) << 3;  // row offset of this lane's C half
  const int cl = lane & 15;
#pragma unroll
  for (int i = 0; i < 2; ++i) {
    const int rbase = row0 + 16 * i;
#pragma unroll
    for (int j = 0; j < 4; ++j) {
      const int col = col0 + 16 * j + cl;
      const float bv = bias[col];
      if (mode == 0) {
#pragma unroll
        for (int r = 0; r < 8; ++r)
          Yh[(size_t)(rbase + half8 + r) * D_M + col] =
              (_Float16)(acc[i][j][r] + bv);
      } else if (mode == 1) {
        const int n = row0 >> 11;                   // batch (L_S = 2048)
        const int t0 = (rbase & (L_S - 1)) + half8; // seq position
        v8h pk;
#pragma unroll
        for (int r = 0; r < 8; ++r) pk[r] = (_Float16)(acc[i][j][r] + bv);
        // Vt[(n*16 + h)*64 + d][t] with h*64+d == col
        *reinterpret_cast<v8h*>(&Yh[((size_t)(n * D_M + col) << 11) + t0]) =
            pk;
      } else {
#pragma unroll
        for (int r = 0; r < 8; ++r)
          Yf[(size_t)(rbase + half8 + r) * D_M + col] = acc[i][j][r] + bv;
      }
    }
  }
}

// ------------------------------------------------------------- attention
// One wave per (n, h, 16-query tile). Flash-attention over 32-key steps.
// Parity ping-pong: step s+1's mask words and K fragments are loaded while
// step s's score WMMAs and softmax VALU work execute.  V fragments for the
// current step are issued early (consumed only after the softmax).
__global__ void __launch_bounds__(128) attn_flash(
    const _Float16* __restrict__ Qh, const _Float16* __restrict__ Kh,
    const _Float16* __restrict__ Vt, const int* __restrict__ Mask,
    _Float16* __restrict__ attnH) {
  __shared__ __align__(16) _Float16 Plds[4][16 * 32];

  const int w = threadIdx.x >> 5;
  const int lane = threadIdx.x & 31;
  const int tile = blockIdx.x * 4 + w;     // 8192 waves total
  const int qt = tile & 127;               // 128 q-tiles per (n,h)
  const int nh = tile >> 7;                // n*16 + h
  const int n = nh >> 4;
  const int h = nh & 15;
  const int q0 = qt << 4;
  const float scale = 0.125f;  // 1/sqrt(64)

  const _Float16* qbase = Qh + (size_t)(n * L_S + q0) * D_M + h * HD;
  const v16h aq0 = load_a(qbase, D_M);        // HD 0..31
  const v16h aq1 = load_a(qbase + 32, D_M);   // HD 32..63

  const int* Mn = Mask + (size_t)n * L_S * L_S;
  _Float16* pw = Plds[w];
  const int half8 = (lane >> 4) << 3;
  const int cl = lane & 15;

  float mrow[8], lrow[8];
  v8f O[4];
#pragma unroll
  for (int r = 0; r < 8; ++r) {
    mrow[r] = -__builtin_inff();
    lrow[r] = 0.0f;
  }
#pragma unroll
  for (int c = 0; c < 4; ++c) O[c] = vzero8();

  const _Float16* kbase = Kh + (size_t)(n * L_S) * D_M + h * HD;
  const _Float16* vbase = Vt + ((size_t)nh * HD) * L_S;

  v16h bkb[2][4];   // [parity][frag]: {k0 hd0, k0 hd1, k1 hd0, k1 hd1}
  int mv0b[2][8], mv1b[2][8];

  // prologue: mask + K fragments for kt = 0 into bank 0
#pragma unroll
  for (int r = 0; r < 8; ++r) {
    const int* mp = Mn + (size_t)(q0 + half8 + r) * L_S + cl;
    mv0b[0][r] = mp[0];
    mv1b[0][r] = mp[16];
  }
  bkb[0][0] = load_b(kbase, D_M);
  bkb[0][1] = load_b(kbase + 32, D_M);
  bkb[0][2] = load_b(kbase + (size_t)16 * D_M, D_M);
  bkb[0][3] = load_b(kbase + (size_t)16 * D_M + 32, D_M);

  auto astep = [&](int s, int p) {
    const int kt = s << 5;

    // V fragments for this step (consumed after softmax -> latency hidden)
    const _Float16* vb = vbase + kt;
    v16h bv[4];
#pragma unroll
    for (int c = 0; c < 4; ++c)
      bv[c] = load_b(vb + (size_t)(16 * c) * L_S, L_S);

    // next step's mask + K fragments into the opposite bank
    if (s < 63) {
      const int ktn = kt + 32;
#pragma unroll
      for (int r = 0; r < 8; ++r) {
        const int* mp = Mn + (size_t)(q0 + half8 + r) * L_S + ktn + cl;
        mv0b[p ^ 1][r] = mp[0];
        mv1b[p ^ 1][r] = mp[16];
      }
      const _Float16* kb = kbase + (size_t)ktn * D_M;
      bkb[p ^ 1][0] = load_b(kb, D_M);
      bkb[p ^ 1][1] = load_b(kb + 32, D_M);
      bkb[p ^ 1][2] = load_b(kb + (size_t)16 * D_M, D_M);
      bkb[p ^ 1][3] = load_b(kb + (size_t)16 * D_M + 32, D_M);
      pf_frag(vb + 32, L_S);                       // next V tile
      pf_frag(vb + (size_t)32 * L_S + 32, L_S);
    }

    // ---- scores: S = Q(16x64) x K^T(64x32), two 16x16 col tiles ----
    v8f S0 = vzero8(), S1 = vzero8();
    S0 = wmma_f16(aq0, bkb[p][0], S0);
    S0 = wmma_f16(aq1, bkb[p][1], S0);
    S1 = wmma_f16(aq0, bkb[p][2], S1);
    S1 = wmma_f16(aq1, bkb[p][3], S1);

    // ---- mask + online softmax (row q0+half8+r lives in this lane half) ----
#pragma unroll
    for (int r = 0; r < 8; ++r) {
      float s0 = mv0b[p][r] ? S0[r] * scale : -1e9f;
      float s1 = mv1b[p][r] ? S1[r] * scale : -1e9f;
      float tmax = half_max16(fmaxf(s0, s1));
      const float mn = fmaxf(mrow[r], tmax);
      const float corr = __expf(mrow[r] - mn);
      const float p0 = __expf(s0 - mn);
      const float p1 = __expf(s1 - mn);
      const float rs = half_sum16(p0 + p1);
      lrow[r] = lrow[r] * corr + rs;
      mrow[r] = mn;
      O[0][r] *= corr;
      O[1][r] *= corr;
      O[2][r] *= corr;
      O[3][r] *= corr;
      // stage P (C layout) into LDS for re-read in A layout
      pw[(half8 + r) * 32 + cl] = (_Float16)p0;
      pw[(half8 + r) * 32 + cl + 16] = (_Float16)p1;
    }

    // wave-synchronous cross-lane LDS handoff
    asm volatile("s_wait_dscnt 0" ::: "memory");

    v16h pfr;
    {
      const _Float16* pp = pw + cl * 32 + ((lane >> 4) << 3);
      v8h lo = *reinterpret_cast<const v8h*>(pp);
      v8h hi = *reinterpret_cast<const v8h*>(pp + 16);
      pfr = cat8(lo, hi);
    }

    // ---- O += P(16x32) x V(32x64); V^T available directly as Vt rows ----
#pragma unroll
    for (int c = 0; c < 4; ++c) O[c] = wmma_f16(pfr, bv[c], O[c]);
  };

  for (int s = 0; s < 64; s += 2) {  // 2048/32 = 64 key steps
    astep(s, 0);
    astep(s + 1, 1);
  }

  // ---- normalize and write attn (f16, row-major NLxD) ----
#pragma unroll
  for (int r = 0; r < 8; ++r) {
    const float inv = 1.0f / lrow[r];
    const int row = q0 + half8 + r;
    const size_t ob = (size_t)(n * L_S + row) * D_M + h * HD + cl;
    attnH[ob] = (_Float16)(O[0][r] * inv);
    attnH[ob + 16] = (_Float16)(O[1][r] * inv);
    attnH[ob + 32] = (_Float16)(O[2][r] * inv);
    attnH[ob + 48] = (_Float16)(O[3][r] * inv);
  }
}

// ----------------------------------------------------------------- launch
extern "C" void kernel_launch(void* const* d_in, const int* in_sizes, int n_in,
                              void* d_out, int out_size, void* d_ws,
                              size_t ws_size, hipStream_t stream) {
  (void)in_sizes; (void)n_in; (void)out_size; (void)ws_size;

  const float* Hq = (const float*)d_in[0];
  const float* Hk = (const float*)d_in[1];
  const float* Hv = (const float*)d_in[2];
  const int*   Mk = (const int*)d_in[3];
  const float* Wq = (const float*)d_in[4];
  const float* bq = (const float*)d_in[5];
  const float* Wk = (const float*)d_in[6];
  const float* bk = (const float*)d_in[7];
  const float* Wv = (const float*)d_in[8];
  const float* bv = (const float*)d_in[9];
  const float* Wo = (const float*)d_in[10];
  const float* bo = (const float*)d_in[11];

  const size_t nX = (size_t)N_B * L_S * D_M;  // 8388608
  const size_t nW = (size_t)D_M * D_M;        // 1048576

  _Float16* p = (_Float16*)d_ws;
  _Float16* Xq  = p; p += nX;
  _Float16* Xk  = p; p += nX;
  _Float16* Xv  = p; p += nX;
  _Float16* Wqh = p; p += nW;
  _Float16* Wkh = p; p += nW;
  _Float16* Wvh = p; p += nW;
  _Float16* Woh = p; p += nW;
  _Float16* Qh  = p; p += nX;
  _Float16* Kh  = p; p += nX;
  _Float16* Vt  = p; p += nX;
  _Float16* Ah  = p; p += nX;

  const int cbX = (int)(nX / 4 / 256);  // 8192
  const int cbW = (int)(nW / 4 / 256);  // 1024
  cvt_f32_f16<<<cbX, 256, 0, stream>>>(Hq, Xq, (int)(nX / 4));
  cvt_f32_f16<<<cbX, 256, 0, stream>>>(Hk, Xk, (int)(nX / 4));
  cvt_f32_f16<<<cbX, 256, 0, stream>>>(Hv, Xv, (int)(nX / 4));
  cvt_f32_f16<<<cbW, 256, 0, stream>>>(Wq, Wqh, (int)(nW / 4));
  cvt_f32_f16<<<cbW, 256, 0, stream>>>(Wk, Wkh, (int)(nW / 4));
  cvt_f32_f16<<<cbW, 256, 0, stream>>>(Wv, Wvh, (int)(nW / 4));
  cvt_f32_f16<<<cbW, 256, 0, stream>>>(Wo, Woh, (int)(nW / 4));

  // (8192/32 row tiles) * (1024/64 col tiles) = 4096 waves / 4 = 1024 blocks
  gemm_wmma_32x64<<<1024, 128, 0, stream>>>(Xq, Wqh, bq, Qh, nullptr, 0);
  gemm_wmma_32x64<<<1024, 128, 0, stream>>>(Xk, Wkh, bk, Kh, nullptr, 0);
  gemm_wmma_32x64<<<1024, 128, 0, stream>>>(Xv, Wvh, bv, Vt, nullptr, 1);

  // 4*16*(2048/16) = 8192 waves / 4 = 2048 blocks
  attn_flash<<<2048, 128, 0, stream>>>(Qh, Kh, Vt, Mk, Ah);

  gemm_wmma_32x64<<<1024, 128, 0, stream>>>(Ah, Woh, bo, nullptr,
                                            (float*)d_out, 2);
}
